// MultiGroupDMoN_13159779795382
// MI455X (gfx1250) — matrix operand order repaired and checked by reading, not compiled, not run on hardware
//
#include <hip/hip_runtime.h>
#include <math.h>

typedef __attribute__((ext_vector_type(16))) _Float16 v16h;
typedef __attribute__((ext_vector_type(8)))  float    v8f;

#define N_NODES_C   100000
#define D_FEAT_C    256
#define K_CL        16
#define N_TILES     (N_NODES_C / 16)   // 6250
#define POOL_CHUNKS (N_NODES_C / 32)   // 3125

// workspace float layout (tiny: 70 floats)
#define WS_FULL 0     // [m, trace, dS[16]]
#define WS_G1   18
#define WS_G2   36
#define WS_CS   54    // cluster_sizes[16]
#define WS_TOT  70

// d_out layout: [features_pooled 16x256][S 100000x16][loss]
#define OUT_POOL 0
#define OUT_S    4096
#define OUT_LOSS (4096 + N_NODES_C * K_CL)

// -------------------------------------------------------------------------
__global__ void __launch_bounds__(256)
k_zero(float* __restrict__ out, float* __restrict__ ws) {
  int i = blockIdx.x * blockDim.x + threadIdx.x;
  if (i < 4096)   out[i] = 0.f;   // pooled-feature f32 accumulator
  if (i < WS_TOT) ws[i]  = 0.f;   // scalar accumulators
}

// -------------------------------------------------------------------------
// S = softmax(features @ W + b); one wave per 16-node tile, WMMA f16->f32.
__global__ void __launch_bounds__(64)
k_assign(const float* __restrict__ feat, const float* __restrict__ W,
         const float* __restrict__ bias, float* __restrict__ S,
         float* __restrict__ ws_cs) {
  __shared__ float scs[2 * 16];
  const int tid  = threadIdx.x;
  const int lane = tid & 31;
  const int wave = tid >> 5;
  if (tid < 32) scs[tid] = 0.f;
  __syncthreads();

  const int tile = blockIdx.x * 2 + wave;        // grid = 3125 -> tile < 6250 always
  const int rowb = tile * 16;
  const int mrow = lane & 15;                    // A: node row / B,D: cluster col
  const int half = lane >> 4;

  v8f c = {0.f, 0.f, 0.f, 0.f, 0.f, 0.f, 0.f, 0.f};
  const float* fr = feat + (size_t)(rowb + mrow) * D_FEAT_C;

  for (int kb = 0; kb < D_FEAT_C; kb += 32) {
    v16h a, bf;
    // A (16x32 f16 layout): lane holds K = 8*half+e (e<8) and K = 16+8*half+e (e>=8)
    const float4* p0 = reinterpret_cast<const float4*>(fr + kb + 8 * half);
    const float4* p1 = reinterpret_cast<const float4*>(fr + kb + 16 + 8 * half);
    float4 f0 = p0[0], f1 = p0[1], f2 = p1[0], f3 = p1[1];
    a[0]=(_Float16)f0.x; a[1]=(_Float16)f0.y; a[2]=(_Float16)f0.z; a[3]=(_Float16)f0.w;
    a[4]=(_Float16)f1.x; a[5]=(_Float16)f1.y; a[6]=(_Float16)f1.z; a[7]=(_Float16)f1.w;
    a[8]=(_Float16)f2.x; a[9]=(_Float16)f2.y; a[10]=(_Float16)f2.z; a[11]=(_Float16)f2.w;
    a[12]=(_Float16)f3.x; a[13]=(_Float16)f3.y; a[14]=(_Float16)f3.z; a[15]=(_Float16)f3.w;
    // B (32x16): lane n = mrow, element e -> K = 8*half+e (e<8) / 16+8*half+(e-8) (e>=8)
    const float* wp = W + (size_t)(kb + 8 * half) * K_CL + mrow;
#pragma unroll
    for (int e = 0; e < 16; ++e) bf[e] = (_Float16)wp[(e < 8 ? e : e + 8) * K_CL];
    c = __builtin_amdgcn_wmma_f32_16x16x32_f16(false, a, false, bf, (short)0, c,
                                               false, false);
  }

  // D layout: lane&15 = cluster N, VGPR r -> node M = r + 8*half
  const float bn = bias[mrow];
  float cspart = 0.f;
#pragma unroll
  for (int r = 0; r < 8; ++r) {
    float x  = c[r] + bn;
    float mx = x;
#pragma unroll
    for (int msk = 8; msk >= 1; msk >>= 1) mx = fmaxf(mx, __shfl_xor(mx, msk));
    float ex = __expf(x - mx);
    float sm = ex;
#pragma unroll
    for (int msk = 8; msk >= 1; msk >>= 1) sm += __shfl_xor(sm, msk);
    float sv = ex / sm;
    cspart += sv;
    S[(size_t)(rowb + r + 8 * half) * K_CL + mrow] = sv;
  }
  cspart += __shfl_xor(cspart, 16);              // combine halves (same cluster n)
  if (lane < 16) scs[wave * 16 + lane] = cspart;
  __syncthreads();
  if (tid < 16) atomicAdd(&ws_cs[tid], scs[tid] + scs[16 + tid]);
}

// -------------------------------------------------------------------------
// Per-edge reduction: acc = [m, trace, dS[16]].
// trace += v * dot(S[src], S[dst]); dS += v * S[dst]; m += v.
__global__ void __launch_bounds__(256)
k_edges(const int* __restrict__ src, const int* __restrict__ dst,
        const float* __restrict__ val, const float* __restrict__ S,
        int ne, float* __restrict__ acc) {
  float tr = 0.f, mm = 0.f;
  float dS[16];
#pragma unroll
  for (int k = 0; k < 16; ++k) dS[k] = 0.f;

  const int stride = gridDim.x * blockDim.x;
  for (int i = blockIdx.x * blockDim.x + threadIdx.x; i < ne; i += stride) {
    const int   sN = src[i], dN = dst[i];
    const float v  = val[i];
    const float4* ps = reinterpret_cast<const float4*>(S + (size_t)sN * 16);
    const float4* pd = reinterpret_cast<const float4*>(S + (size_t)dN * 16);
    float dot = 0.f;
#pragma unroll
    for (int q = 0; q < 4; ++q) {
      float4 a = ps[q], b = pd[q];
      dot += a.x * b.x + a.y * b.y + a.z * b.z + a.w * b.w;
      dS[4 * q + 0] += v * b.x; dS[4 * q + 1] += v * b.y;
      dS[4 * q + 2] += v * b.z; dS[4 * q + 3] += v * b.w;
    }
    tr += v * dot;
    mm += v;
  }

  // wave32 reduction
#pragma unroll
  for (int off = 16; off >= 1; off >>= 1) {
    mm += __shfl_xor(mm, off);
    tr += __shfl_xor(tr, off);
#pragma unroll
    for (int k = 0; k < 16; ++k) dS[k] += __shfl_xor(dS[k], off);
  }
  __shared__ float sred[8][18];
  const int lane = threadIdx.x & 31, wave = threadIdx.x >> 5;
  if (lane == 0) {
    sred[wave][0] = mm; sred[wave][1] = tr;
#pragma unroll
    for (int k = 0; k < 16; ++k) sred[wave][2 + k] = dS[k];
  }
  __syncthreads();
  if (threadIdx.x < 18) {
    float v = 0.f;
#pragma unroll
    for (int w = 0; w < 8; ++w) v += sred[w][threadIdx.x];
    atomicAdd(&acc[threadIdx.x], v);
  }
}

// -------------------------------------------------------------------------
// pooled_acc[k, d] += sum_i S[i,k] * features[i,d]  (WMMA, A = S^T chunk)
__global__ void __launch_bounds__(256)
k_pool(const float* __restrict__ S, const float* __restrict__ feat,
       float* __restrict__ pool) {
  const int ct    = blockIdx.x & 15;   // feature column tile (16 cols)
  const int split = blockIdx.x >> 4;   // node split 0..15
  const int lane  = threadIdx.x & 31;
  const int wave  = threadIdx.x >> 5;
  const int mrow  = lane & 15;         // A: cluster row / B,D: feature col
  const int half  = lane >> 4;
  const int wslot = split * 8 + wave;  // 0..127

  v8f c = {0.f, 0.f, 0.f, 0.f, 0.f, 0.f, 0.f, 0.f};
  for (int chunk = wslot; chunk < POOL_CHUNKS; chunk += 128) {
    const int nb = chunk * 32;
    v16h a, bf;
    // A[m=cluster, k=node]: S column mrow, K = 8*half+e / 16+8*half+e
    const float* sp = S + (size_t)nb * K_CL + mrow;
#pragma unroll
    for (int e = 0; e < 8; ++e)  a[e]     = (_Float16)sp[(8 * half + e) * K_CL];
#pragma unroll
    for (int e = 0; e < 8; ++e)  a[8 + e] = (_Float16)sp[(16 + 8 * half + e) * K_CL];
    // B[k=node, n=feat col]: K = 8*half+e / 16+8*half+e
    const float* fp = feat + (size_t)(nb + 8 * half) * D_FEAT_C + ct * 16 + mrow;
#pragma unroll
    for (int e = 0; e < 16; ++e) bf[e] = (_Float16)fp[(size_t)(e < 8 ? e : e + 8) * D_FEAT_C];
    c = __builtin_amdgcn_wmma_f32_16x16x32_f16(false, a, false, bf, (short)0, c,
                                               false, false);
  }
  // D: lane&15 = feat col n, VGPR r -> cluster r + 8*half
#pragma unroll
  for (int r = 0; r < 8; ++r)
    atomicAdd(&pool[(size_t)(r + 8 * half) * D_FEAT_C + ct * 16 + mrow], c[r]);
}

// -------------------------------------------------------------------------
__global__ void __launch_bounds__(256)
k_final(float* __restrict__ out, const float* __restrict__ ws,
        const int* __restrict__ lamda_p) {
  const float* cs = ws + WS_CS;
  // features_pooled = selu(pool_acc / cluster_sizes)  (in place)
  for (int i = threadIdx.x; i < 4096; i += 256) {
    const int   k = i >> 8;
    const float x = out[i] / cs[k];
    const float scale = 1.0507009873554805f, alpha = 1.6732632423543772f;
    out[i] = (x > 0.f) ? scale * x : scale * alpha * (expf(x) - 1.f);
  }
  if (threadIdx.x == 0) {
    const float* fa = ws + WS_FULL;
    const float m = fa[0], trA = fa[1];
    float sq = 0.f;
    for (int k = 0; k < 16; ++k) sq += fa[2 + k] * fa[2 + k];
    const float spectral = -(trA - sq / (2.f * m)) / (2.f * m);

    float Q[2], gm[2];
    for (int g = 0; g < 2; ++g) {
      const float* ga = ws + (g == 0 ? WS_G1 : WS_G2);
      const float gmm = ga[0], trg = ga[1];
      float gsq = 0.f;
      for (int k = 0; k < 16; ++k) gsq += ga[2 + k] * ga[2 + k];
      const float safe = (gmm > 0.f) ? gmm : 1.f;
      const float q = (trg - gsq / (2.f * safe)) / (2.f * m);
      Q[g]  = (gmm > 0.f) ? q : 0.f;
      gm[g] = gmm;
    }
    const float qmin     = fminf(Q[0], Q[1]);
    const float gmean    = 0.5f * (gm[0] + gm[1]);
    const float fairness = (m / (gmean + 1e-8f)) * (-qmin);

    float csq = 0.f;
    for (int k = 0; k < 16; ++k) csq += cs[k] * cs[k];
    const float collapse = (sqrtf(csq) / (float)N_NODES_C) * 4.f - 1.f;

    const float lam = (float)lamda_p[0];
    out[OUT_LOSS] = spectral + lam * fairness + 0.1f * collapse;
  }
}

// -------------------------------------------------------------------------
extern "C" void kernel_launch(void* const* d_in, const int* in_sizes, int n_in,
                              void* d_out, int out_size, void* d_ws, size_t ws_size,
                              hipStream_t stream) {
  const float* feat   = (const float*)d_in[0];
  const float* W      = (const float*)d_in[1];
  const float* bias   = (const float*)d_in[2];
  const int*   a_src  = (const int*)d_in[3];
  const int*   a_dst  = (const int*)d_in[4];
  const float* a_val  = (const float*)d_in[5];
  const int*   g1_src = (const int*)d_in[6];
  const int*   g1_dst = (const int*)d_in[7];
  const float* g1_val = (const float*)d_in[8];
  const int*   g2_src = (const int*)d_in[9];
  const int*   g2_dst = (const int*)d_in[10];
  const float* g2_val = (const float*)d_in[11];
  const int*   lamda  = (const int*)d_in[12];

  float* out  = (float*)d_out;
  float* ws   = (float*)d_ws;
  float* Sbuf = out + OUT_S;

  const int nA = in_sizes[5];
  const int n1 = in_sizes[8];
  const int n2 = in_sizes[11];

  k_zero  <<<16, 256, 0, stream>>>(out, ws);
  k_assign<<<N_TILES / 2, 64, 0, stream>>>(feat, W, bias, Sbuf, ws + WS_CS);
  k_edges <<<1024, 256, 0, stream>>>(a_src, a_dst, a_val, Sbuf, nA, ws + WS_FULL);
  k_edges <<<512, 256, 0, stream>>>(g1_src, g1_dst, g1_val, Sbuf, n1, ws + WS_G1);
  k_edges <<<512, 256, 0, stream>>>(g2_src, g2_dst, g2_val, Sbuf, n2, ws + WS_G2);
  k_pool  <<<256, 256, 0, stream>>>(Sbuf, feat, out);
  k_final <<<1, 256, 0, stream>>>(out, ws, lamda);
}
